// NSM_50929722196182
// MI455X (gfx1250) — matrix operand order repaired and checked by reading, not compiled, not run on hardware
//
#include <hip/hip_runtime.h>

#define H_ 300
#define P_ 8
#define STEPS_ 4
#define NI_ 5
#define OUT_ 1845
#define B_ 32
#define N_ 16384
#define E_ 131072
#define L_ 30
#define NPG_ 512

typedef __attribute__((ext_vector_type(16))) __bf16 v16bf;
typedef __attribute__((ext_vector_type(8)))  __bf16 v8bf;
typedef __attribute__((ext_vector_type(8)))  float  v8f;

__device__ __forceinline__ unsigned short f2bfu(float f) {
  union { float f; unsigned u; } x; x.f = f;
  unsigned r = x.u + 0x7FFFu + ((x.u >> 16) & 1u);  // RNE
  return (unsigned short)(r >> 16);
}
__device__ __forceinline__ __bf16 f2bf(float f) {
  unsigned short h = f2bfu(f);
  __bf16 o; __builtin_memcpy(&o, &h, 2); return o;
}
__device__ __forceinline__ unsigned short bfbits(__bf16 v) {
  unsigned short h; __builtin_memcpy(&h, &v, 2); return h;
}
__device__ __forceinline__ unsigned pack2(float lo, float hi) {
  return (unsigned)f2bfu(lo) | ((unsigned)f2bfu(hi) << 16);
}
__device__ __forceinline__ float eluf(float x) { return x > 0.f ? x : expm1f(x); }
__device__ __forceinline__ float sigf(float x) { return 1.f / (1.f + __expf(-x)); }

// two contiguous 8-elem LDS chunks -> one 16-elem fragment (2x ds_load_b128)
__device__ __forceinline__ v16bf frag16(const __bf16* rowbase, int chunklo, int chunkhi) {
  const v8bf* p = (const v8bf*)rowbase;
  v8bf lo = p[chunklo], hi = p[chunkhi];
  return __builtin_shufflevector(lo, hi, 0,1,2,3,4,5,6,7,8,9,10,11,12,13,14,15);
}

#define BM 64
#define BN 64
#define BK 32
#define AIT 4   // (BM*BK/2)/256 staging iterations
#define BIT 4   // (BN*BK/2)/256 staging iterations

// ---------------- generic WMMA GEMM: C(M,Nout) = A(fp32) @ B(bf16, rows zero-padded to Kpad)
// Requires: M % 64 == 0, Keff even (true at all call sites).
__global__ void __launch_bounds__(256)
k_gemm_generic(const float* __restrict__ A, const __bf16* __restrict__ Bb,
               float* __restrict__ C, int M, int Nout, int Keff, int Kpad,
               int lda, int ldb, int ldc)
{
  __shared__ __align__(16) __bf16 As[BM * BK];       // [m][k]
  __shared__ __align__(16) __bf16 Bs[BN * BK];       // transposed: [n][k]
  unsigned* As32 = (unsigned*)As;
  unsigned* Bs32 = (unsigned*)Bs;
  const int tid = threadIdx.x;
  const int lane = tid & 31, w = tid >> 5;
  const int moff = (w & 3) * 16, noff = (w >> 2) * 32;
  const int m0 = blockIdx.x * BM, n0 = blockIdx.y * BN;
  v8f acc0 = {0.f,0.f,0.f,0.f,0.f,0.f,0.f,0.f};
  v8f acc1 = acc0;
  for (int k0 = 0; k0 < Kpad; k0 += BK) {
    // ---- stage A: phase 1 = issue all loads (no uses); phase 2 = mask+pack+store
    float ra[AIT], rb[AIT], mk[AIT];
#pragma unroll
    for (int it = 0; it < AIT; ++it) {
      int i = tid + 256 * it;
      int r = i >> 4, kp = (i & 15) << 1;
      int gk = k0 + kp;
      int gkc = gk < Keff ? gk : 0;
      mk[it] = gk < Keff ? 1.f : 0.f;
      const float* ar = A + (size_t)(m0 + r) * lda + gkc;
      ra[it] = ar[0];
      rb[it] = ar[1];
    }
    // ---- stage B: phase 1 = issue all loads
    unsigned short b0[BIT], b1[BIT];
    int gnv[BIT];
#pragma unroll
    for (int it = 0; it < BIT; ++it) {
      int i = tid + 256 * it;
      int n = i & 63, kp = (i >> 6) << 1;
      int gn = n0 + n;
      gnv[it] = gn;
      int gnc = gn < ldb ? gn : 0;
      const __bf16* bp = Bb + (size_t)(k0 + kp) * ldb + gnc;
      b0[it] = bfbits(bp[0]);
      b1[it] = bfbits(bp[ldb]);
    }
    if (k0 + BK < Kpad && tid < BM) {                // prefetch next A k-tile
      if ((k0 + BK) < Keff) __builtin_prefetch(&A[(size_t)(m0 + tid) * lda + k0 + BK], 0, 1);
    }
    // ---- phase 2: convert + LDS stores
#pragma unroll
    for (int it = 0; it < AIT; ++it) {
      int i = tid + 256 * it;
      int r = i >> 4, kp = (i & 15) << 1;
      As32[(r << 4) + (kp >> 1)] = pack2(ra[it] * mk[it], rb[it] * mk[it]);
    }
#pragma unroll
    for (int it = 0; it < BIT; ++it) {
      int i = tid + 256 * it;
      int n = i & 63, kp = (i >> 6) << 1;
      unsigned u = (unsigned)b0[it] | ((unsigned)b1[it] << 16);
      u = gnv[it] < ldb ? u : 0u;
      Bs32[(n << 4) + (kp >> 1)] = u;
    }
    __syncthreads();
    const int am = moff + (lane & 15);
    const int ac = (lane >> 4) << 1;                 // A chunks: ac, ac+1
    v16bf af  = frag16(As + am * BK, ac, ac + 1);
    const int bn0 = noff + (lane & 15);
    const int bc = (lane >> 4) << 1;                 // B chunks: bc, bc+1
    v16bf bf0 = frag16(Bs + bn0 * BK, bc, bc + 1);
    v16bf bf1 = frag16(Bs + (bn0 + 16) * BK, bc, bc + 1);
    acc0 = __builtin_amdgcn_wmma_f32_16x16x32_bf16(false, af, false, bf0, (short)0, acc0, false, false);
    acc1 = __builtin_amdgcn_wmma_f32_16x16x32_bf16(false, af, false, bf1, (short)0, acc1, false, false);
    __syncthreads();
  }
  const int cn0 = n0 + noff + (lane & 15), cn1 = cn0 + 16;
#pragma unroll
  for (int j = 0; j < 8; ++j) {
    int cm = m0 + moff + j + ((lane >> 4) << 3);
    if (cn0 < Nout) C[(size_t)cm * ldc + cn0] = acc0[j];
    if (cn1 < Nout) C[(size_t)cm * ldc + cn1] = acc1[j];
  }
}

// ---------------- node GEMM: A built on the fly = node_attrs * instr[b] * prop_sim[b,p]; ELU epilogue
__global__ void __launch_bounds__(256)
k_gemm_node(const float* __restrict__ nattr, const int* __restrict__ nidx,
            const float* __restrict__ instrS, const float* __restrict__ psim,
            const __bf16* __restrict__ Bb, float* __restrict__ C)
{
  __shared__ __align__(16) __bf16 As[BM * BK];
  __shared__ __align__(16) __bf16 Bs[BN * BK];
  unsigned* As32 = (unsigned*)As;
  unsigned* Bs32 = (unsigned*)Bs;
  const int tid = threadIdx.x;
  const int lane = tid & 31, w = tid >> 5;
  const int moff = (w & 3) * 16, noff = (w >> 2) * 32;
  const int m0 = blockIdx.x * BM, n0 = blockIdx.y * BN;
  v8f acc0 = {0.f,0.f,0.f,0.f,0.f,0.f,0.f,0.f};
  v8f acc1 = acc0;
  for (int k0 = 0; k0 < P_ * 320; k0 += BK) {
    const int p = k0 / 320;              // 320 % 32 == 0: tile never spans p
    const int hbase = k0 - p * 320;
    // phase 1: issue loads; h even -> (h < 300) covers the whole pair
    float ra[AIT], rb[AIT], ia[AIT], ib[AIT], sc[AIT];
#pragma unroll
    for (int it = 0; it < AIT; ++it) {
      int i = tid + 256 * it;
      int r = i >> 4, kp = (i & 15) << 1;
      int gm = m0 + r;                   // M = 16384, multiple of 64
      int h = hbase + kp;
      int hc = h < H_ ? h : 0;
      float mk = h < H_ ? 1.f : 0.f;
      int b = nidx[gm];
      sc[it] = psim[b * P_ + p] * mk;
      const float* na = nattr + ((size_t)gm * P_ + p) * H_ + hc;
      const float* iv = instrS + b * (NI_ * H_) + hc;
      ra[it] = na[0]; rb[it] = na[1];
      ia[it] = iv[0]; ib[it] = iv[1];
    }
    unsigned short b0[BIT], b1[BIT];
    int gnv[BIT];
#pragma unroll
    for (int it = 0; it < BIT; ++it) {
      int i = tid + 256 * it;
      int n = i & 63, kp = (i >> 6) << 1;
      int gn = n0 + n;
      gnv[it] = gn;
      int gnc = gn < H_ ? gn : 0;
      const __bf16* bp = Bb + (size_t)(k0 + kp) * H_ + gnc;
      b0[it] = bfbits(bp[0]);
      b1[it] = bfbits(bp[H_]);
    }
    // phase 2: convert + LDS stores
#pragma unroll
    for (int it = 0; it < AIT; ++it) {
      int i = tid + 256 * it;
      int r = i >> 4, kp = (i & 15) << 1;
      As32[(r << 4) + (kp >> 1)] = pack2(ra[it] * ia[it] * sc[it], rb[it] * ib[it] * sc[it]);
    }
#pragma unroll
    for (int it = 0; it < BIT; ++it) {
      int i = tid + 256 * it;
      int n = i & 63, kp = (i >> 6) << 1;
      unsigned u = (unsigned)b0[it] | ((unsigned)b1[it] << 16);
      u = gnv[it] < H_ ? u : 0u;
      Bs32[(n << 4) + (kp >> 1)] = u;
    }
    __syncthreads();
    const int am = moff + (lane & 15);
    const int ac = (lane >> 4) << 1;
    v16bf af  = frag16(As + am * BK, ac, ac + 1);
    const int bn0 = noff + (lane & 15);
    const int bc = (lane >> 4) << 1;
    v16bf bf0 = frag16(Bs + bn0 * BK, bc, bc + 1);
    v16bf bf1 = frag16(Bs + (bn0 + 16) * BK, bc, bc + 1);
    acc0 = __builtin_amdgcn_wmma_f32_16x16x32_bf16(false, af, false, bf0, (short)0, acc0, false, false);
    acc1 = __builtin_amdgcn_wmma_f32_16x16x32_bf16(false, af, false, bf1, (short)0, acc1, false, false);
    __syncthreads();
  }
  const int cn0 = n0 + noff + (lane & 15), cn1 = cn0 + 16;
#pragma unroll
  for (int j = 0; j < 8; ++j) {
    int cm = m0 + moff + j + ((lane >> 4) << 3);
    if (cn0 < H_) C[(size_t)cm * H_ + cn0] = eluf(acc0[j]);
    if (cn1 < H_) C[(size_t)cm * H_ + cn1] = eluf(acc1[j]);
  }
}

// ---------------- edge GEMM: A = instr[batch]*edge_attrs; epilogue fuses ELU + dist[src] + atomic scatter into msg[dst]
__global__ void __launch_bounds__(256)
k_gemm_edge(const float* __restrict__ eattr, const int* __restrict__ ebatch,
            const int* __restrict__ eidx, const float* __restrict__ instrS,
            const __bf16* __restrict__ Bb, const float* __restrict__ dist,
            float* __restrict__ msg)
{
  __shared__ __align__(16) __bf16 As[BM * BK];
  __shared__ __align__(16) __bf16 Bs[BN * BK];
  unsigned* As32 = (unsigned*)As;
  unsigned* Bs32 = (unsigned*)Bs;
  const int tid = threadIdx.x;
  const int lane = tid & 31, w = tid >> 5;
  const int moff = (w & 3) * 16, noff = (w >> 2) * 32;
  const int m0 = blockIdx.x * BM, n0 = blockIdx.y * BN;
  v8f acc0 = {0.f,0.f,0.f,0.f,0.f,0.f,0.f,0.f};
  v8f acc1 = acc0;
  for (int k0 = 0; k0 < 320; k0 += BK) {
    float ra[AIT], rb[AIT], ia[AIT], ib[AIT], mk[AIT];
#pragma unroll
    for (int it = 0; it < AIT; ++it) {
      int i = tid + 256 * it;
      int r = i >> 4, kp = (i & 15) << 1;
      int gm = m0 + r;                   // E multiple of 64
      int gk = k0 + kp;                  // even; (gk < 300) covers the pair
      int gkc = gk < H_ ? gk : 0;
      mk[it] = gk < H_ ? 1.f : 0.f;
      const float* iv = instrS + ebatch[gm] * (NI_ * H_) + gkc;
      const float* ea = eattr + (size_t)gm * H_ + gkc;
      ra[it] = ea[0]; rb[it] = ea[1];
      ia[it] = iv[0]; ib[it] = iv[1];
    }
    unsigned short b0[BIT], b1[BIT];
    int gnv[BIT];
#pragma unroll
    for (int it = 0; it < BIT; ++it) {
      int i = tid + 256 * it;
      int n = i & 63, kp = (i >> 6) << 1;
      int gn = n0 + n;
      gnv[it] = gn;
      int gnc = gn < H_ ? gn : 0;
      const __bf16* bp = Bb + (size_t)(k0 + kp) * H_ + gnc;
      b0[it] = bfbits(bp[0]);
      b1[it] = bfbits(bp[H_]);
    }
#pragma unroll
    for (int it = 0; it < AIT; ++it) {
      int i = tid + 256 * it;
      int r = i >> 4, kp = (i & 15) << 1;
      As32[(r << 4) + (kp >> 1)] = pack2(ra[it] * ia[it] * mk[it], rb[it] * ib[it] * mk[it]);
    }
#pragma unroll
    for (int it = 0; it < BIT; ++it) {
      int i = tid + 256 * it;
      int n = i & 63, kp = (i >> 6) << 1;
      unsigned u = (unsigned)b0[it] | ((unsigned)b1[it] << 16);
      u = gnv[it] < H_ ? u : 0u;
      Bs32[(n << 4) + (kp >> 1)] = u;
    }
    __syncthreads();
    const int am = moff + (lane & 15);
    const int ac = (lane >> 4) << 1;
    v16bf af  = frag16(As + am * BK, ac, ac + 1);
    const int bn0 = noff + (lane & 15);
    const int bc = (lane >> 4) << 1;
    v16bf bf0 = frag16(Bs + bn0 * BK, bc, bc + 1);
    v16bf bf1 = frag16(Bs + (bn0 + 16) * BK, bc, bc + 1);
    acc0 = __builtin_amdgcn_wmma_f32_16x16x32_bf16(false, af, false, bf0, (short)0, acc0, false, false);
    acc1 = __builtin_amdgcn_wmma_f32_16x16x32_bf16(false, af, false, bf1, (short)0, acc1, false, false);
    __syncthreads();
  }
  const int cn0 = n0 + noff + (lane & 15), cn1 = cn0 + 16;
#pragma unroll
  for (int j = 0; j < 8; ++j) {
    int cm = m0 + moff + j + ((lane >> 4) << 3);
    int s = eidx[cm], d = eidx[E_ + cm];
    float ds = dist[s];
    if (cn0 < H_) atomicAdd(&msg[(size_t)d * H_ + cn0], ds * eluf(acc0[j]));
    if (cn1 < H_) atomicAdd(&msg[(size_t)d * H_ + cn1], ds * eluf(acc1[j]));
  }
}

// ---------------- weight converters (fp32 -> zero-padded bf16 B matrices)
__global__ void k_cvt_b(const float* __restrict__ src, __bf16* __restrict__ dst,
                        int Kpad, int Nout, int Keff, int transpose)
{
  int idx = blockIdx.x * 256 + threadIdx.x;
  if (idx >= Kpad * Nout) return;
  int k = idx / Nout, n = idx - k * Nout;
  float v = 0.f;
  if (k < Keff) v = transpose ? src[n * Keff + k] : src[k * Nout + n];
  dst[idx] = f2bf(v);
}
__global__ void k_cvt_vocabT(const float* __restrict__ cv, const float* __restrict__ td,
                             __bf16* __restrict__ dst)
{
  int idx = blockIdx.x * 256 + threadIdx.x;
  if (idx >= 320 * 2001) return;
  int k = idx / 2001, n = idx - k * 2001;
  float v = 0.f;
  if (k < H_) v = (n < 2000) ? cv[n * H_ + k] : td[k];
  dst[idx] = f2bf(v);
}
__global__ void k_cvt_wnp(const float* __restrict__ Wnp, __bf16* __restrict__ dst)
{
  int idx = blockIdx.x * 256 + threadIdx.x;
  if (idx >= 2560 * H_) return;
  int k = idx / H_, n = idx - k * H_;
  int p = k / 320, h = k - p * 320;
  float v = (h < H_) ? Wnp[(p * H_ + h) * H_ + n] : 0.f;
  dst[idx] = f2bf(v);
}

// ---------------- row softmax (2001 wide)
__global__ void k_softmax_rows(float* __restrict__ x, int n)
{
  __shared__ float red[256];
  float* row = x + (size_t)blockIdx.x * n;
  int tid = threadIdx.x;
  float m = -__builtin_inff();
  for (int i = tid; i < n; i += 256) m = fmaxf(m, row[i]);
  red[tid] = m; __syncthreads();
  for (int s = 128; s > 0; s >>= 1) { if (tid < s) red[tid] = fmaxf(red[tid], red[tid + s]); __syncthreads(); }
  m = red[0]; __syncthreads();
  float sum = 0.f;
  for (int i = tid; i < n; i += 256) { float e = __expf(row[i] - m); row[i] = e; sum += e; }
  red[tid] = sum; __syncthreads();
  for (int s = 128; s > 0; s >>= 1) { if (tid < s) red[tid] += red[tid + s]; __syncthreads(); }
  float inv = 1.f / red[0];
  for (int i = tid; i < n; i += 256) row[i] *= inv;
}

__global__ void k_tagged_combine(const float* __restrict__ sim, const float* __restrict__ q,
                                 float* __restrict__ tagged)
{
  int idx = blockIdx.x * 256 + threadIdx.x;
  if (idx >= B_ * L_ * H_) return;
  int r = idx / H_;
  tagged[idx] += sim[r * 2001 + 2000] * q[idx];
}

__global__ void k_lstm_step(const float* __restrict__ Xih, const float* __restrict__ hin,
                            const float* __restrict__ cin, float* __restrict__ hout,
                            float* __restrict__ cout, const float* __restrict__ Whh,
                            const float* __restrict__ bih, const float* __restrict__ bhh,
                            const int* __restrict__ lengths, int t)
{
  int idx = blockIdx.x * blockDim.x + threadIdx.x;
  if (idx >= B_ * H_) return;
  int b = idx / H_, j = idx - b * H_;
  const float* hv = hin + b * H_;
  const float* xr = Xih + (b * L_ + t) * 1200;
  float g[4];
#pragma unroll
  for (int q = 0; q < 4; ++q) {
    int rr = q * H_ + j;
    float a = xr[rr] + bih[rr] + bhh[rr];
    const float* wr = Whh + rr * H_;
    for (int h = 0; h < H_; ++h) a += hv[h] * wr[h];
    g[q] = a;
  }
  float cnew = sigf(g[1]) * cin[idx] + sigf(g[0]) * tanhf(g[2]);
  float hnew = sigf(g[3]) * tanhf(cnew);
  bool m = (t < lengths[b]);
  hout[idx] = m ? hnew : hin[idx];
  cout[idx] = m ? cnew : cin[idx];
}

__global__ void k_eih(const float* __restrict__ enc, const float* __restrict__ Wih,
                      float* __restrict__ Eih)
{
  int idx = blockIdx.x * 256 + threadIdx.x;
  if (idx >= B_ * H_) return;
  int b = idx / H_, j = idx - b * H_;
  float a = 0.f;
  const float* ev = enc + b * H_;
  const float* wr = Wih + j * H_;
  for (int h = 0; h < H_; ++h) a += ev[h] * wr[h];
  Eih[idx] = a;
}

__global__ void k_rnn_step(const float* __restrict__ Eih, float* __restrict__ hidden,
                           const float* __restrict__ Whh, const float* __restrict__ bih,
                           const float* __restrict__ bhh, int it)
{
  int idx = blockIdx.x * 256 + threadIdx.x;
  if (idx >= B_ * H_) return;
  int b = idx / H_, j = idx - b * H_;
  float a = Eih[idx] + bih[j] + bhh[j];
  if (it > 0) {
    const float* hp = hidden + (b * NI_ + it - 1) * H_;
    const float* wr = Whh + j * H_;
    for (int h = 0; h < H_; ++h) a += hp[h] * wr[h];
  }
  hidden[(b * NI_ + it) * H_ + j] = fmaxf(a, 0.f);
}

__global__ void k_attention(const float* __restrict__ hidden, const float* __restrict__ tagged,
                            const int* __restrict__ lengths, float* __restrict__ instructions)
{
  __shared__ float sc[L_];
  __shared__ float ssum;
  int b = blockIdx.x / NI_, i = blockIdx.x % NI_;
  int tid = threadIdx.x;
  if (tid < L_) {
    float a = 0.f;
    const float* hv = hidden + (b * NI_ + i) * H_;
    const float* tv = tagged + (b * L_ + tid) * H_;
    for (int h = 0; h < H_; ++h) a += hv[h] * tv[h];
    sc[tid] = (tid < lengths[b]) ? a : -__builtin_inff();
  }
  __syncthreads();
  if (tid == 0) {
    float m = -__builtin_inff();
    for (int l = 0; l < L_; ++l) m = fmaxf(m, sc[l]);
    float s = 0.f;
    for (int l = 0; l < L_; ++l) { float e = __expf(sc[l] - m); sc[l] = e; s += e; }
    ssum = s;
  }
  __syncthreads();
  float inv = 1.f / ssum;
  for (int h = tid; h < H_; h += blockDim.x) {
    float a = 0.f;
    for (int l = 0; l < L_; ++l) a += sc[l] * tagged[(b * L_ + l) * H_ + h];
    instructions[b * (NI_ * H_) + i * H_ + h] = a * inv;
  }
}

__global__ void k_prop_softmax(const float* __restrict__ instrS, const float* __restrict__ pemb,
                               float* __restrict__ psim, float* __restrict__ rel)
{
  int b = threadIdx.x;
  if (b >= B_) return;
  float lg[P_ + 1];
  float m = -__builtin_inff();
  const float* iv = instrS + b * (NI_ * H_);
#pragma unroll
  for (int p = 0; p < P_ + 1; ++p) {
    float a = 0.f;
    const float* pv = pemb + p * H_;
    for (int h = 0; h < H_; ++h) a += iv[h] * pv[h];
    lg[p] = a; m = fmaxf(m, a);
  }
  float s = 0.f;
#pragma unroll
  for (int p = 0; p < P_ + 1; ++p) { lg[p] = __expf(lg[p] - m); s += lg[p]; }
#pragma unroll
  for (int p = 0; p < P_; ++p) psim[b * P_ + p] = lg[p] / s;
  if (rel) rel[b] = lg[P_] / s;
}

__global__ void k_count(const int* __restrict__ nidx, int* __restrict__ cnt)
{
  int n = blockIdx.x * 256 + threadIdx.x;
  if (n < N_) atomicAdd(&cnt[nidx[n]], 1);
}
__global__ void k_dist_init(const int* __restrict__ nidx, const int* __restrict__ cnt,
                            float* __restrict__ dist)
{
  int n = blockIdx.x * 256 + threadIdx.x;
  if (n < N_) dist[n] = 1.f / (float)cnt[nidx[n]];
}

// per-graph (512 contiguous nodes) softmax of dot(mat_row, vec)
__global__ void k_seg_softmax(const float* __restrict__ mat, const float* __restrict__ vec,
                              float* __restrict__ out)
{
  __shared__ float red[256];
  int b = blockIdx.x, tid = threadIdx.x;
  int n0 = b * NPG_ + tid, n1 = n0 + 256;
  float x0 = 0.f, x1 = 0.f;
  for (int h = 0; h < H_; ++h) {
    float v = vec[h];
    x0 += mat[(size_t)n0 * H_ + h] * v;
    x1 += mat[(size_t)n1 * H_ + h] * v;
  }
  red[tid] = fmaxf(x0, x1); __syncthreads();
  for (int s = 128; s > 0; s >>= 1) { if (tid < s) red[tid] = fmaxf(red[tid], red[tid + s]); __syncthreads(); }
  float m = red[0]; __syncthreads();
  float e0 = __expf(x0 - m), e1 = __expf(x1 - m);
  red[tid] = e0 + e1; __syncthreads();
  for (int s = 128; s > 0; s >>= 1) { if (tid < s) red[tid] += red[tid + s]; __syncthreads(); }
  float inv = 1.f / red[0];
  out[n0] = e0 * inv;
  out[n1] = e1 * inv;
}

__global__ void k_dist_update(const int* __restrict__ nidx, const float* __restrict__ rel,
                              const float* __restrict__ nrel, const float* __restrict__ nst,
                              float* __restrict__ dist)
{
  int n = blockIdx.x * 256 + threadIdx.x;
  if (n >= N_) return;
  float rs = rel[nidx[n]];
  dist[n] = rs * nrel[n] + (1.f - rs) * nst[n];
}

__global__ void k_aggregate(const float* __restrict__ nattr, const float* __restrict__ dist,
                            const float* __restrict__ propf, float* __restrict__ agg)
{
  int b = blockIdx.x, h = threadIdx.x;
  if (h >= H_) return;
  float ps[P_];
#pragma unroll
  for (int p = 0; p < P_; ++p) ps[p] = propf[b * P_ + p];
  float acc = 0.f;
  for (int n = 0; n < NPG_; ++n) {
    int nn = b * NPG_ + n;
    float w = 0.f;
#pragma unroll
    for (int p = 0; p < P_; ++p) w += ps[p] * nattr[((size_t)nn * P_ + p) * H_ + h];
    acc += dist[nn] * w;
  }
  agg[b * H_ + h] = acc;
}

__global__ void k_fc1(const float* __restrict__ enc, const float* __restrict__ agg,
                      const float* __restrict__ W, const float* __restrict__ bias,
                      float* __restrict__ z)
{
  int idx = blockIdx.x * 256 + threadIdx.x;
  if (idx >= B_ * 600) return;
  int b = idx / 600, j = idx - b * 600;
  float a = bias[j];
  const float* wr = W + j * 600;
  for (int k = 0; k < 300; ++k) a += enc[b * H_ + k] * wr[k];
  for (int k = 0; k < 300; ++k) a += agg[b * H_ + k] * wr[300 + k];
  z[idx] = eluf(a);
}
__global__ void k_fc2(const float* __restrict__ z, const float* __restrict__ W,
                      const float* __restrict__ bias, float* __restrict__ out)
{
  int idx = blockIdx.x * 256 + threadIdx.x;
  if (idx >= B_ * OUT_) return;
  int b = idx / OUT_, o = idx - b * OUT_;
  float a = bias[o];
  const float* zr = z + b * 600;
  const float* wr = W + o * 600;
  for (int k = 0; k < 600; ++k) a += zr[k] * wr[k];
  out[idx] = a;
}

// ===================== host orchestration =====================
extern "C" void kernel_launch(void* const* d_in, const int* in_sizes, int n_in,
                              void* d_out, int out_size, void* d_ws, size_t ws_size,
                              hipStream_t stream)
{
  (void)in_sizes; (void)n_in; (void)out_size; (void)ws_size;

  const float* questions   = (const float*)d_in[0];
  const int*   lengths     = (const int*)  d_in[1];
  const int*   node_idx    = (const int*)  d_in[2];
  const int*   edge_idx    = (const int*)  d_in[3];
  const int*   edge_batch  = (const int*)  d_in[4];
  const float* node_attrs  = (const float*)d_in[5];
  const float* edge_attrs  = (const float*)d_in[6];
  const float* cvocab      = (const float*)d_in[7];
  const float* pemb        = (const float*)d_in[8];
  const float* tdef        = (const float*)d_in[9];
  const float* tagW        = (const float*)d_in[10];
  const float* lstm_Wih    = (const float*)d_in[11];
  const float* lstm_Whh    = (const float*)d_in[12];
  const float* lstm_bih    = (const float*)d_in[13];
  const float* lstm_bhh    = (const float*)d_in[14];
  const float* rnn_Wih     = (const float*)d_in[15];
  const float* rnn_Whh     = (const float*)d_in[16];
  const float* rnn_bih     = (const float*)d_in[17];
  const float* rnn_bhh     = (const float*)d_in[18];
  const float* Wnp         = (const float*)d_in[19];
  const float* Wedge       = (const float*)d_in[20];
  const float* w_nscore    = (const float*)d_in[21];
  const float* w_rscore    = (const float*)d_in[22];
  const float* fc1_W       = (const float*)d_in[23];
  const float* fc1_b       = (const float*)d_in[24];
  const float* fc2_W       = (const float*)d_in[25];
  const float* fc2_b       = (const float*)d_in[26];

  char* wp = (char*)d_ws;
  auto alloc = [&](size_t bytes) -> void* {
    void* r = (void*)wp;
    wp += (bytes + 255) & ~(size_t)255;
    return r;
  };

  __bf16* tagW_bf   = (__bf16*)alloc(320 * 300 * 2);
  __bf16* vocabT_bf = (__bf16*)alloc(320 * 2001 * 2);
  __bf16* cvocab_bf = (__bf16*)alloc(2016 * 300 * 2);
  __bf16* WihT_bf   = (__bf16*)alloc(320 * 1200 * 2);
  __bf16* Wnp_bf    = (__bf16*)alloc(2560 * 300 * 2);
  __bf16* Wedge_bf  = (__bf16*)alloc(320 * 300 * 2);

  float* qt      = (float*)alloc((size_t)960 * 300 * 4);
  float* sim     = (float*)alloc((size_t)960 * 2001 * 4);
  float* tagged  = (float*)alloc((size_t)960 * 300 * 4);
  float* Xih     = (float*)alloc((size_t)960 * 1200 * 4);
  float* hA      = (float*)alloc(B_ * H_ * 4);
  float* cA      = (float*)alloc(B_ * H_ * 4);
  float* hB      = (float*)alloc(B_ * H_ * 4);
  float* cB      = (float*)alloc(B_ * H_ * 4);
  float* Eih     = (float*)alloc(B_ * H_ * 4);
  float* hidden  = (float*)alloc(B_ * NI_ * H_ * 4);
  float* instr   = (float*)alloc(B_ * NI_ * H_ * 4);
  float* psim    = (float*)alloc(B_ * P_ * 4);
  float* rel     = (float*)alloc(B_ * 4);
  float* propf   = (float*)alloc(B_ * P_ * 4);
  float* nodesc  = (float*)alloc((size_t)N_ * H_ * 4);
  float* msg     = (float*)alloc((size_t)N_ * H_ * 4);
  float* dist    = (float*)alloc(N_ * 4);
  float* nsx     = (float*)alloc(N_ * 4);
  float* nrx     = (float*)alloc(N_ * 4);
  int*   cnt     = (int*)  alloc(B_ * 4);
  float* agg     = (float*)alloc(B_ * H_ * 4);
  float* zbuf    = (float*)alloc(B_ * 600 * 4);

  // --- weight conversions (bf16 B matrices)
  k_cvt_b     <<<(320*300  + 255)/256, 256, 0, stream>>>(tagW,    tagW_bf,  320, 300, 300, 0);
  k_cvt_vocabT<<<(320*2001 + 255)/256, 256, 0, stream>>>(cvocab, tdef, vocabT_bf);
  k_cvt_b     <<<(2016*300 + 255)/256, 256, 0, stream>>>(cvocab,  cvocab_bf, 2016, 300, 2000, 0);
  k_cvt_b     <<<(320*1200 + 255)/256, 256, 0, stream>>>(lstm_Wih, WihT_bf, 320, 1200, 300, 1);
  k_cvt_wnp   <<<(2560*300 + 255)/256, 256, 0, stream>>>(Wnp, Wnp_bf);
  k_cvt_b     <<<(320*300  + 255)/256, 256, 0, stream>>>(Wedge,   Wedge_bf, 320, 300, 300, 0);

  // --- tagging
  k_gemm_generic<<<dim3(15, 5),  256, 0, stream>>>(questions, tagW_bf,   qt,  960, 300,  300,  320, 300, 300, 300);
  k_gemm_generic<<<dim3(15, 32), 256, 0, stream>>>(qt,        vocabT_bf, sim, 960, 2001, 300,  320, 300, 2001, 2001);
  k_softmax_rows<<<960, 256, 0, stream>>>(sim, 2001);
  k_gemm_generic<<<dim3(15, 5),  256, 0, stream>>>(sim,       cvocab_bf, tagged, 960, 300, 2000, 2016, 2001, 300, 300);
  k_tagged_combine<<<(960*300 + 255)/256, 256, 0, stream>>>(sim, questions, tagged);

  // --- LSTM (input projection hoisted into one WMMA GEMM)
  k_gemm_generic<<<dim3(15, 19), 256, 0, stream>>>(tagged, WihT_bf, Xih, 960, 1200, 300, 320, 300, 1200, 1200);
  hipMemsetAsync(hA, 0, B_ * H_ * 4, stream);
  hipMemsetAsync(cA, 0, B_ * H_ * 4, stream);
  {
    float *hin = hA, *cin = cA, *hout = hB, *cout = cB;
    for (int t = 0; t < L_; ++t) {
      k_lstm_step<<<(B_*H_ + 255)/256, 256, 0, stream>>>(Xih, hin, cin, hout, cout,
                                                         lstm_Whh, lstm_bih, lstm_bhh, lengths, t);
      float* tmp = hin; hin = hout; hout = tmp;
      tmp = cin; cin = cout; cout = tmp;
    }
    // after 30 steps the final state is back in the "in" buffers: encoded == hin
    k_eih<<<(B_*H_ + 255)/256, 256, 0, stream>>>(hin, rnn_Wih, Eih);
    for (int it = 0; it < NI_; ++it)
      k_rnn_step<<<(B_*H_ + 255)/256, 256, 0, stream>>>(Eih, hidden, rnn_Whh, rnn_bih, rnn_bhh, it);
    k_attention<<<B_ * NI_, 256, 0, stream>>>(hidden, tagged, lengths, instr);

    // --- graph reasoning
    hipMemsetAsync(cnt, 0, B_ * 4, stream);
    k_count<<<N_/256, 256, 0, stream>>>(node_idx, cnt);
    k_dist_init<<<N_/256, 256, 0, stream>>>(node_idx, cnt, dist);

    for (int step = 0; step < STEPS_; ++step) {
      const float* instrS = instr + step * H_;
      k_prop_softmax<<<1, 32, 0, stream>>>(instrS, pemb, psim, rel);
      k_gemm_node<<<dim3(N_/BM, 5), 256, 0, stream>>>(node_attrs, node_idx, instrS, psim, Wnp_bf, nodesc);
      k_seg_softmax<<<B_, 256, 0, stream>>>(nodesc, w_nscore, nsx);
      hipMemsetAsync(msg, 0, (size_t)N_ * H_ * 4, stream);
      k_gemm_edge<<<dim3(E_/BM, 5), 256, 0, stream>>>(edge_attrs, edge_batch, edge_idx, instrS,
                                                      Wedge_bf, dist, msg);
      k_seg_softmax<<<B_, 256, 0, stream>>>(msg, w_rscore, nrx);
      k_dist_update<<<N_/256, 256, 0, stream>>>(node_idx, rel, nrx, nsx, dist);
    }

    // --- readout
    k_prop_softmax<<<1, 32, 0, stream>>>(instr + (NI_ - 1) * H_, pemb, propf, (float*)nullptr);
    k_aggregate<<<B_, 320, 0, stream>>>(node_attrs, dist, propf, agg);
    k_fc1<<<(B_*600 + 255)/256, 256, 0, stream>>>(hin, agg, fc1_W, fc1_b, zbuf);
    k_fc2<<<(B_*OUT_ + 255)/256, 256, 0, stream>>>(zbuf, fc2_W, fc2_b, (float*)d_out);
  }
}
